// Llama4StyleMoE_71640054497666
// MI455X (gfx1250) — compile-verified, matrix operand order checked by reading, not run on hardware
//
#include <hip/hip_runtime.h>

typedef __bf16 bf16;
typedef __attribute__((ext_vector_type(16))) __bf16 v16bf_t;
typedef __attribute__((ext_vector_type(8)))  float  v8f_t;

#define T_TOK 4096
#define HDIM  1024
#define NEXP  16
#define FDIM  1024
#define SFDIM 4096

// ---- workspace layout (bytes) ----
#define EIDX_OFF   0u            // int[T]
#define SCALE_OFF  16384u        // float[T]
#define CNT_OFF    32768u        // int[16]
#define CNT2_OFF   32832u        // int[16]
#define OFFS_OFF   32896u        // int[16]
#define PERM_OFF   33024u        // int[T]
#define XS_OFF     65536u        // bf16[T*H]   scaled tokens (routed path)
#define XB_OFF     (XS_OFF  + 2u*T_TOK*HDIM)   // bf16[T*H]  plain tokens (shared path)
#define ACTR_OFF   (XB_OFF  + 2u*T_TOK*HDIM)   // bf16[T*F]  routed swiglu activations
#define ACTS_OFF   (ACTR_OFF+ 2u*T_TOK*FDIM)   // bf16[T*SF] shared swiglu activations

// ---------------------------------------------------------------------------
// WMMA helpers
// ---------------------------------------------------------------------------
__device__ __forceinline__ v8f_t wmma_bf16(v16bf_t a, v16bf_t b, v8f_t c) {
  return __builtin_amdgcn_wmma_f32_16x16x32_bf16(false, a, false, b, (short)0, c,
                                                 false, false);
}

// LDS tile is stored "row-major over the lane index" with 32 bf16 per row.
// A operand: lane m = lane&15, half g = lane>>4; elements 2v,2v+1 hold
// K = 8g+2v (v<4) / 16+8g+2(v-4) (v>=4)  — matches ISA 16-bit A layout.
// B operand uses the symmetric layout with n = lane&15.
__device__ __forceinline__ v16bf_t frag_ld(const bf16* rowbase, int g) {
  union { v16bf_t v; unsigned u[8]; } r;
  const unsigned* q = (const unsigned*)rowbase;  // 2 bf16 per dword
  int b = 4 * g;
#pragma unroll
  for (int v = 0; v < 4; v++) {
    r.u[v]     = q[b + v];       // K = 8g .. 8g+7
    r.u[4 + v] = q[8 + b + v];   // K = 16+8g .. 16+8g+7
  }
  return r.v;
}

__device__ __forceinline__ float silu_f(float x) {
  return x / (1.0f + __expf(-x));
}

// ---------------------------------------------------------------------------
// 0) zero counters
// ---------------------------------------------------------------------------
__global__ void k_zero_meta(int* cnt, int* cnt2) {
  if (threadIdx.x < NEXP) { cnt[threadIdx.x] = 0; cnt2[threadIdx.x] = 0; }
}

// ---------------------------------------------------------------------------
// 1) router: one wave per token; 16 dot products of length H, top-1 + sigmoid
// ---------------------------------------------------------------------------
__global__ __launch_bounds__(128) void k_router(const float* __restrict__ x,
                                                const float* __restrict__ rw,
                                                int* __restrict__ eidx,
                                                float* __restrict__ scale,
                                                int* __restrict__ cnt) {
  int wave = threadIdx.x >> 5;
  int lane = threadIdx.x & 31;
  int t = blockIdx.x * 4 + wave;
  const float* xr = x + (size_t)t * HDIM;
  float best = -3.4e38f;
  int   bi   = 0;
#pragma unroll 1
  for (int e = 0; e < NEXP; e++) {
    const float* wr = rw + (size_t)e * HDIM;
    float s = 0.f;
    for (int h = lane; h < HDIM; h += 32) s += xr[h] * wr[h];
#pragma unroll
    for (int off = 16; off > 0; off >>= 1) s += __shfl_xor(s, off, 32);
    if (s > best) { best = s; bi = e; }   // ties -> lowest index, like top_k
  }
  if (lane == 0) {
    eidx[t]  = bi;
    scale[t] = 1.0f / (1.0f + __expf(-best));
    atomicAdd(&cnt[bi], 1);
  }
}

// ---------------------------------------------------------------------------
// 2) exclusive scan of 16 counts
// ---------------------------------------------------------------------------
__global__ void k_scan(const int* __restrict__ cnt, int* __restrict__ offs) {
  if (threadIdx.x == 0) {
    int a = 0;
    for (int e = 0; e < NEXP; e++) { offs[e] = a; a += cnt[e]; }
  }
}

// ---------------------------------------------------------------------------
// 3) scatter tokens into per-expert buckets
// ---------------------------------------------------------------------------
__global__ void k_scatter(const int* __restrict__ eidx, const int* __restrict__ offs,
                          int* __restrict__ cnt2, int* __restrict__ perm) {
  int t = blockIdx.x * blockDim.x + threadIdx.x;
  if (t < T_TOK) {
    int e = eidx[t];
    int p = atomicAdd(&cnt2[e], 1);
    perm[offs[e] + p] = t;
  }
}

// ---------------------------------------------------------------------------
// 4) convert hidden -> bf16 (plain) and bf16*sigmoid (routed input)
// ---------------------------------------------------------------------------
__global__ __launch_bounds__(256) void k_convert(const float* __restrict__ x,
                                                 const float* __restrict__ scale,
                                                 bf16* __restrict__ xs,
                                                 bf16* __restrict__ xb) {
  size_t i = ((size_t)blockIdx.x * blockDim.x + threadIdx.x) * 4;
  size_t t = i >> 10;  // H == 1024
  float  s = scale[t];
  float4 v = *(const float4*)(x + i);
  xb[i + 0] = (bf16)v.x; xb[i + 1] = (bf16)v.y;
  xb[i + 2] = (bf16)v.z; xb[i + 3] = (bf16)v.w;
  xs[i + 0] = (bf16)(v.x * s); xs[i + 1] = (bf16)(v.y * s);
  xs[i + 2] = (bf16)(v.z * s); xs[i + 3] = (bf16)(v.w * s);
}

// ---------------------------------------------------------------------------
// 5) shared expert: gate+up fused GEMM + SwiGLU -> act_s[T, SF] (bf16)
//    A: xb[T,H] bf16.  B: shared_gate_w / shared_up_w [SF,H] fp32 (N-major).
//    Block tile 64x128, 8 waves, K step 32.
// ---------------------------------------------------------------------------
__global__ __launch_bounds__(256) void k_gemm1_shared(const bf16* __restrict__ xb,
                                                      const float* __restrict__ wg,
                                                      const float* __restrict__ wu,
                                                      bf16* __restrict__ act) {
  __shared__ __align__(16) bf16 As[64 * 32];
  __shared__ __align__(16) bf16 Bg[128 * 32];
  __shared__ __align__(16) bf16 Bu[128 * 32];
  int tid = threadIdx.x;
  int t0 = blockIdx.y * 64;
  int f0 = blockIdx.x * 128;
  int wv = tid >> 5, lane = tid & 31, g = lane >> 4, lr = lane & 15;
  int ms = (wv & 3) * 16;
  int ns = (wv >> 2) * 64;
  v8f_t accg[4] = {};
  v8f_t accu[4] = {};
#pragma unroll 1
  for (int k0 = 0; k0 < HDIM; k0 += 32) {
    __syncthreads();
    {  // A tile 64x32 bf16: 8 elems (16B) per thread
      int row = tid >> 2, col = (tid & 3) * 8;
      *(uint4*)&As[row * 32 + col] =
          *(const uint4*)&xb[(size_t)(t0 + row) * HDIM + k0 + col];
    }
    {  // B tiles 128x32, fp32 -> bf16 on the fly (N-major, direct)
      int f = tid >> 1, hc = (tid & 1) * 16;
      const float* pg = wg + (size_t)(f0 + f) * HDIM + k0 + hc;
      const float* pu = wu + (size_t)(f0 + f) * HDIM + k0 + hc;
      bf16* dg = &Bg[f * 32 + hc];
      bf16* du = &Bu[f * 32 + hc];
#pragma unroll
      for (int j = 0; j < 16; j += 4) {
        float4 vg = *(const float4*)(pg + j);
        float4 vu = *(const float4*)(pu + j);
        dg[j] = (bf16)vg.x; dg[j+1] = (bf16)vg.y; dg[j+2] = (bf16)vg.z; dg[j+3] = (bf16)vg.w;
        du[j] = (bf16)vu.x; du[j+1] = (bf16)vu.y; du[j+2] = (bf16)vu.z; du[j+3] = (bf16)vu.w;
      }
      if (k0 + 32 < HDIM) {  // global_prefetch_b8 for next weight tile
        __builtin_prefetch(pg + 32, 0, 1);
        __builtin_prefetch(pu + 32, 0, 1);
      }
    }
    __syncthreads();
    v16bf_t a = frag_ld(&As[(ms + lr) * 32], g);
#pragma unroll
    for (int i = 0; i < 4; i++) {
      v16bf_t bg = frag_ld(&Bg[(ns + i * 16 + lr) * 32], g);
      accg[i] = wmma_bf16(a, bg, accg[i]);
      v16bf_t bu = frag_ld(&Bu[(ns + i * 16 + lr) * 32], g);
      accu[i] = wmma_bf16(a, bu, accu[i]);
    }
  }
#pragma unroll
  for (int i = 0; i < 4; i++) {
#pragma unroll
    for (int r = 0; r < 8; r++) {
      int m = r + 8 * g;
      float o = accu[i][r] * silu_f(accg[i][r]);
      act[(size_t)(t0 + ms + m) * SFDIM + f0 + ns + i * 16 + lr] = (bf16)o;
    }
  }
}

// ---------------------------------------------------------------------------
// 6) shared expert down: out[T,H] = act_s[T,SF] @ shared_down_w[H,SF]^T
//    (overwrites d_out; routed path atomically adds afterwards)
// ---------------------------------------------------------------------------
__global__ __launch_bounds__(256) void k_gemm2_shared(const bf16* __restrict__ act,
                                                      const float* __restrict__ wd,
                                                      float* __restrict__ out) {
  __shared__ __align__(16) bf16 As[64 * 32];
  __shared__ __align__(16) bf16 Bs[128 * 32];
  int tid = threadIdx.x;
  int t0 = blockIdx.y * 64;
  int h0 = blockIdx.x * 128;
  int wv = tid >> 5, lane = tid & 31, g = lane >> 4, lr = lane & 15;
  int ms = (wv & 3) * 16;
  int ns = (wv >> 2) * 64;
  v8f_t acc[4] = {};
#pragma unroll 1
  for (int k0 = 0; k0 < SFDIM; k0 += 32) {
    __syncthreads();
    {
      int row = tid >> 2, col = (tid & 3) * 8;
      *(uint4*)&As[row * 32 + col] =
          *(const uint4*)&act[(size_t)(t0 + row) * SFDIM + k0 + col];
    }
    {
      int hh = tid >> 1, c = (tid & 1) * 16;
      const float* p = wd + (size_t)(h0 + hh) * SFDIM + k0 + c;
      bf16* d = &Bs[hh * 32 + c];
#pragma unroll
      for (int j = 0; j < 16; j += 4) {
        float4 v = *(const float4*)(p + j);
        d[j] = (bf16)v.x; d[j+1] = (bf16)v.y; d[j+2] = (bf16)v.z; d[j+3] = (bf16)v.w;
      }
      if (k0 + 32 < SFDIM) __builtin_prefetch(p + 32, 0, 1);
    }
    __syncthreads();
    v16bf_t a = frag_ld(&As[(ms + lr) * 32], g);
#pragma unroll
    for (int i = 0; i < 4; i++) {
      v16bf_t b = frag_ld(&Bs[(ns + i * 16 + lr) * 32], g);
      acc[i] = wmma_bf16(a, b, acc[i]);
    }
  }
#pragma unroll
  for (int i = 0; i < 4; i++) {
#pragma unroll
    for (int r = 0; r < 8; r++) {
      int m = r + 8 * g;
      out[(size_t)(t0 + ms + m) * HDIM + h0 + ns + i * 16 + lr] = acc[i][r];
    }
  }
}

// ---------------------------------------------------------------------------
// 7) routed gate+up + SwiGLU over gathered tokens of one expert
//    W = gate_up_proj[e] : [H, 2F] fp32 (K-major -> transpose into LDS)
// ---------------------------------------------------------------------------
__global__ __launch_bounds__(256) void k_gemm1_routed(const bf16* __restrict__ xs,
                                                      const float* __restrict__ gup,
                                                      const int* __restrict__ perm,
                                                      const int* __restrict__ offs,
                                                      const int* __restrict__ cnt,
                                                      bf16* __restrict__ actr) {
  int e = blockIdx.z;
  int count = cnt[e];
  int m0 = blockIdx.y * 64;
  if (m0 >= count) return;
  int off = offs[e];
  int f0 = blockIdx.x * 128;
  const float* W = gup + (size_t)e * HDIM * (2 * FDIM);

  __shared__ __align__(16) bf16 As[64 * 32];
  __shared__ __align__(16) bf16 Bg[128 * 32];
  __shared__ __align__(16) bf16 Bu[128 * 32];
  __shared__ int toks[64];

  int tid = threadIdx.x;
  if (tid < 64) {
    int rr = m0 + tid;
    toks[tid] = perm[off + (rr < count ? rr : count - 1)];
  }
  int wv = tid >> 5, lane = tid & 31, g = lane >> 4, lr = lane & 15;
  int ms = (wv & 3) * 16;
  int ns = (wv >> 2) * 64;
  v8f_t accg[4] = {};
  v8f_t accu[4] = {};
#pragma unroll 1
  for (int k0 = 0; k0 < HDIM; k0 += 32) {
    __syncthreads();
    {  // gathered A tile
      int row = tid >> 2, col = (tid & 3) * 8;
      *(uint4*)&As[row * 32 + col] =
          *(const uint4*)&xs[(size_t)toks[row] * HDIM + k0 + col];
    }
    {  // B tiles: rows k (h), cols f contiguous -> transpose into [n][k]
      int kk = tid >> 3, fc = (tid & 7) * 16;
      const float* pg = W + (size_t)(k0 + kk) * (2 * FDIM) + f0 + fc;
      const float* pu = pg + FDIM;
#pragma unroll
      for (int j = 0; j < 16; j += 4) {
        float4 vg = *(const float4*)(pg + j);
        float4 vu = *(const float4*)(pu + j);
        Bg[(fc + j + 0) * 32 + kk] = (bf16)vg.x;
        Bg[(fc + j + 1) * 32 + kk] = (bf16)vg.y;
        Bg[(fc + j + 2) * 32 + kk] = (bf16)vg.z;
        Bg[(fc + j + 3) * 32 + kk] = (bf16)vg.w;
        Bu[(fc + j + 0) * 32 + kk] = (bf16)vu.x;
        Bu[(fc + j + 1) * 32 + kk] = (bf16)vu.y;
        Bu[(fc + j + 2) * 32 + kk] = (bf16)vu.z;
        Bu[(fc + j + 3) * 32 + kk] = (bf16)vu.w;
      }
      if (k0 + 32 < HDIM) __builtin_prefetch(pg + 32 * (2 * FDIM), 0, 1);
    }
    __syncthreads();
    v16bf_t a = frag_ld(&As[(ms + lr) * 32], g);
#pragma unroll
    for (int i = 0; i < 4; i++) {
      v16bf_t bg = frag_ld(&Bg[(ns + i * 16 + lr) * 32], g);
      accg[i] = wmma_bf16(a, bg, accg[i]);
      v16bf_t bu = frag_ld(&Bu[(ns + i * 16 + lr) * 32], g);
      accu[i] = wmma_bf16(a, bu, accu[i]);
    }
  }
#pragma unroll
  for (int i = 0; i < 4; i++) {
#pragma unroll
    for (int r = 0; r < 8; r++) {
      int m = r + 8 * g;
      int rr = m0 + ms + m;
      if (rr < count) {
        float o = accu[i][r] * silu_f(accg[i][r]);
        actr[(size_t)(off + rr) * FDIM + f0 + ns + i * 16 + lr] = (bf16)o;
      }
    }
  }
}

// ---------------------------------------------------------------------------
// 8) routed down: atomically add down_proj[e] result into d_out at perm rows
//    W = down_proj[e] : [F, H] fp32 (K-major -> transpose into LDS)
// ---------------------------------------------------------------------------
__global__ __launch_bounds__(256) void k_gemm2_routed(const bf16* __restrict__ actr,
                                                      const float* __restrict__ dwn,
                                                      const int* __restrict__ perm,
                                                      const int* __restrict__ offs,
                                                      const int* __restrict__ cnt,
                                                      float* __restrict__ out) {
  int e = blockIdx.z;
  int count = cnt[e];
  int m0 = blockIdx.y * 64;
  if (m0 >= count) return;
  int off = offs[e];
  int h0 = blockIdx.x * 128;
  const float* W = dwn + (size_t)e * FDIM * HDIM;

  __shared__ __align__(16) bf16 As[64 * 32];
  __shared__ __align__(16) bf16 Bs[128 * 32];
  __shared__ int toks[64];

  int tid = threadIdx.x;
  if (tid < 64) {
    int rr = m0 + tid;
    toks[tid] = (rr < count) ? perm[off + rr] : 0;
  }
  int wv = tid >> 5, lane = tid & 31, g = lane >> 4, lr = lane & 15;
  int ms = (wv & 3) * 16;
  int ns = (wv >> 2) * 64;
  v8f_t acc[4] = {};
#pragma unroll 1
  for (int k0 = 0; k0 < FDIM; k0 += 32) {
    __syncthreads();
    {
      int row = tid >> 2, col = (tid & 3) * 8;
      int rr = m0 + row;
      int src = off + (rr < count ? rr : count - 1);
      *(uint4*)&As[row * 32 + col] =
          *(const uint4*)&actr[(size_t)src * FDIM + k0 + col];
    }
    {
      int kk = tid >> 3, hc = (tid & 7) * 16;
      const float* p = W + (size_t)(k0 + kk) * HDIM + h0 + hc;
#pragma unroll
      for (int j = 0; j < 16; j += 4) {
        float4 v = *(const float4*)(p + j);
        Bs[(hc + j + 0) * 32 + kk] = (bf16)v.x;
        Bs[(hc + j + 1) * 32 + kk] = (bf16)v.y;
        Bs[(hc + j + 2) * 32 + kk] = (bf16)v.z;
        Bs[(hc + j + 3) * 32 + kk] = (bf16)v.w;
      }
      if (k0 + 32 < FDIM) __builtin_prefetch(p + 32 * HDIM, 0, 1);
    }
    __syncthreads();
    v16bf_t a = frag_ld(&As[(ms + lr) * 32], g);
#pragma unroll
    for (int i = 0; i < 4; i++) {
      v16bf_t b = frag_ld(&Bs[(ns + i * 16 + lr) * 32], g);
      acc[i] = wmma_bf16(a, b, acc[i]);
    }
  }
#pragma unroll
  for (int i = 0; i < 4; i++) {
#pragma unroll
    for (int r = 0; r < 8; r++) {
      int m = r + 8 * g;
      int rr = m0 + ms + m;
      if (rr < count) {
        int t = toks[ms + m];
        atomicAdd(&out[(size_t)t * HDIM + h0 + ns + i * 16 + lr], acc[i][r]);
      }
    }
  }
}

// ---------------------------------------------------------------------------
extern "C" void kernel_launch(void* const* d_in, const int* in_sizes, int n_in,
                              void* d_out, int out_size, void* d_ws, size_t ws_size,
                              hipStream_t stream) {
  const float* x   = (const float*)d_in[0];  // [B,S,H]
  const float* rw  = (const float*)d_in[1];  // [N,H]
  const float* gup = (const float*)d_in[2];  // [N,H,2F]
  const float* dwn = (const float*)d_in[3];  // [N,F,H]
  const float* sg  = (const float*)d_in[4];  // [SF,H]
  const float* su  = (const float*)d_in[5];  // [SF,H]
  const float* sd  = (const float*)d_in[6];  // [H,SF]
  float* out = (float*)d_out;

  char* ws = (char*)d_ws;
  int*   eidx  = (int*)(ws + EIDX_OFF);
  float* scale = (float*)(ws + SCALE_OFF);
  int*   cnt   = (int*)(ws + CNT_OFF);
  int*   cnt2  = (int*)(ws + CNT2_OFF);
  int*   offs  = (int*)(ws + OFFS_OFF);
  int*   perm  = (int*)(ws + PERM_OFF);
  bf16*  xs    = (bf16*)(ws + XS_OFF);
  bf16*  xb    = (bf16*)(ws + XB_OFF);
  bf16*  actr  = (bf16*)(ws + ACTR_OFF);
  bf16*  acts  = (bf16*)(ws + ACTS_OFF);

  k_zero_meta<<<1, 64, 0, stream>>>(cnt, cnt2);
  k_router<<<T_TOK / 4, 128, 0, stream>>>(x, rw, eidx, scale, cnt);
  k_scan<<<1, 32, 0, stream>>>(cnt, offs);
  k_scatter<<<T_TOK / 256, 256, 0, stream>>>(eidx, offs, cnt2, perm);
  k_convert<<<(T_TOK * HDIM / 4) / 256, 256, 0, stream>>>(x, scale, xs, xb);

  k_gemm1_shared<<<dim3(SFDIM / 128, T_TOK / 64), 256, 0, stream>>>(xb, sg, su, acts);
  k_gemm2_shared<<<dim3(HDIM / 128, T_TOK / 64), 256, 0, stream>>>(acts, sd, out);

  k_gemm1_routed<<<dim3(FDIM / 128, T_TOK / 64, NEXP), 256, 0, stream>>>(
      xs, gup, perm, offs, cnt, actr);
  k_gemm2_routed<<<dim3(HDIM / 128, T_TOK / 64, NEXP), 256, 0, stream>>>(
      actr, dwn, perm, offs, cnt, out);
}
